// ExhaustiveBiaffineNERDecoder_19988777795986
// MI455X (gfx1250) — matrix operand order repaired and checked by reading, not compiled, not run on hardware
//
#include <hip/hip_runtime.h>
#include <hip/hip_bf16.h>

// ---- problem constants (from the reference) ----
#define NLAB   24
#define LDIM   256
#define DIM    1024
#define NB     8
#define LL     512
#define OUTCH  (NLAB * LDIM * 2)   // 12288
#define NTOK   (NB * LL)           // 4096
#define NEGV   (-10000.0f)

#define LDS_PITCH 40               // 32 bf16 + 8 pad = 80B row stride (16B aligned,
                                   // 20-bank stride => conflict-free 16B ds loads)

typedef __bf16 bf16_t;
typedef __attribute__((ext_vector_type(16))) __bf16 v16bf;
typedef __attribute__((ext_vector_type(8)))  __bf16 v8bf;
typedef __attribute__((ext_vector_type(4)))  __bf16 v4bf;
typedef __attribute__((ext_vector_type(8)))  float  v8f;
typedef __attribute__((ext_vector_type(4)))  float  v4f;

// ---------------------------------------------------------------------------
// CDNA5 async global->LDS copy (ASYNCcnt-tracked, no VGPR round trip).
// LDS aperture rule: generic LDS address low 32 bits == LDS byte offset.
// ---------------------------------------------------------------------------
__device__ __forceinline__ void async_copy_b128(const bf16_t* __restrict__ g,
                                                bf16_t* __restrict__ l)
{
    unsigned           laddr = (unsigned)(unsigned long long)l;
    unsigned long long gaddr = (unsigned long long)g;
    asm volatile("global_load_async_to_lds_b128 %0, %1, off"
                 :: "v"(laddr), "v"(gaddr) : "memory");
}

__device__ __forceinline__ void wait_async()
{
    asm volatile("s_wait_asynccnt 0x0" ::: "memory");
}

// ---------------------------------------------------------------------------
// Stage one 128-row x 32-col bf16 tile into LDS (512 x 16B segments,
// 2 async b128 per thread with 256 threads).
// ---------------------------------------------------------------------------
template<int LDG>
__device__ __forceinline__ void stage_tile(const bf16_t* __restrict__ g, int k0,
                                           bf16_t* __restrict__ s, int tid)
{
    #pragma unroll
    for (int j = 0; j < 2; ++j) {
        const int l   = tid + 256 * j;
        const int row = l >> 2;
        const int seg = l & 3;
        async_copy_b128(g + (long)row * LDG + k0 + seg * 8,
                        s + row * LDS_PITCH + seg * 8);
    }
}

// ---------------------------------------------------------------------------
// Double-buffered LDS-tiled GEMM mainloop.
// Block tile 128(M) x 128(N); 8 waves in a 4x2 grid, each wave owns a
// 32(M) x 64(N) register tile = 2x4 WMMA accumulators. Per K=32 slice each
// wave issues 8 v_wmma_f32_16x16x32_bf16 from LDS-resident fragments while
// the next slice streams in asynchronously.
//
// Fragment layouts follow the CDNA5 16-bit WMMA VGPR spec:
//   A (16x32): lane m=lane&15 -> row m; kh=lane>>4 selects K-half:
//              VGPR0..3 = K kh*8..kh*8+7, VGPR4..7 = K 16+kh*8..16+kh*8+7
//   B (32x16): lane n=lane&15 -> column n; same K order as A
// ---------------------------------------------------------------------------
template<int KTOT, int LDG>
__device__ __forceinline__ void gemm_tiles(const bf16_t* __restrict__ Ag,
                                           const bf16_t* __restrict__ Bg,
                                           bf16_t (*sA)[128 * LDS_PITCH],
                                           bf16_t (*sB)[128 * LDS_PITCH],
                                           v8f c[2][4])
{
    const int tid  = threadIdx.x;
    const int lane = tid & 31;
    const int wid  = tid >> 5;
    const int wm   = wid & 3;   // M block (32 rows)
    const int wn   = wid >> 2;  // N block (64 cols)
    const int m    = lane & 15;
    const int kh   = lane >> 4;

    stage_tile<LDG>(Ag, 0, sA[0], tid);
    stage_tile<LDG>(Bg, 0, sB[0], tid);
    wait_async();
    __syncthreads();

    int buf = 0;
    for (int k0 = 0; k0 < KTOT; k0 += 32) {
        if (k0 + 32 < KTOT) {
            stage_tile<LDG>(Ag, k0 + 32, sA[buf ^ 1], tid);
            stage_tile<LDG>(Bg, k0 + 32, sB[buf ^ 1], tid);
        }

        v16bf a[2], b[4];
        #pragma unroll
        for (int i = 0; i < 2; ++i) {
            const bf16_t* p = sA[buf] + (wm * 32 + i * 16 + m) * LDS_PITCH + kh * 8;
            v8bf lo = *(const v8bf*)p;
            v8bf hi = *(const v8bf*)(p + 16);
            #pragma unroll
            for (int q = 0; q < 8; ++q) { a[i][q] = lo[q]; a[i][q + 8] = hi[q]; }
        }
        #pragma unroll
        for (int j = 0; j < 4; ++j) {
            const bf16_t* p = sB[buf] + (wn * 64 + j * 16 + m) * LDS_PITCH + kh * 8;
            v8bf lo = *(const v8bf*)p;
            v8bf hi = *(const v8bf*)(p + 16);
            #pragma unroll
            for (int q = 0; q < 8; ++q) { b[j][q] = lo[q]; b[j][q + 8] = hi[q]; }
        }

        #pragma unroll
        for (int i = 0; i < 2; ++i)
            #pragma unroll
            for (int j = 0; j < 4; ++j)
                c[i][j] = __builtin_amdgcn_wmma_f32_16x16x32_bf16(
                              false, a[i], false, b[j], (short)0, c[i][j], false, false);

        wait_async();
        __syncthreads();
        buf ^= 1;
    }
}

// ---------------------------------------------------------------------------
// f32 -> bf16 conversion (vectorized x4)
// ---------------------------------------------------------------------------
__global__ __launch_bounds__(256)
void cvt_f32_to_bf16(const float* __restrict__ src, bf16_t* __restrict__ dst, long n)
{
    long i = ((long)blockIdx.x * blockDim.x + threadIdx.x) * 4;
    if (i < n) {
        v4f v = *(const v4f*)(src + i);
        v4bf o;
        #pragma unroll
        for (int j = 0; j < 4; ++j) o[j] = (__bf16)v[j];
        *(v4bf*)(dst + i) = o;
    }
}

// ---------------------------------------------------------------------------
// Stage 1: FFN GEMM  C[t, o] = relu( feat[t,:] . w[o,:] + b_ff[o] )
// 128x128 block tiles, K=1024. Epilogue de-interleaves even/odd channels
// into start/end bf16 buffers laid out [b, label, s, d] (K-major for stage 2).
// ---------------------------------------------------------------------------
__global__ __launch_bounds__(256)
void ffn_wmma(const bf16_t* __restrict__ feat,     // [NTOK, DIM] bf16
              const bf16_t* __restrict__ w,        // [OUTCH, DIM] bf16
              const float*  __restrict__ b_ff,     // [OUTCH]
              bf16_t* __restrict__ start_buf,      // [NB, NLAB, LL, LDIM]
              bf16_t* __restrict__ end_buf)        // [NB, NLAB, LL, LDIM]
{
    __shared__ bf16_t sA[2][128 * LDS_PITCH];
    __shared__ bf16_t sB[2][128 * LDS_PITCH];

    const int bm     = blockIdx.x % (NTOK / 128);  // 32 token blocks
    const int bn     = blockIdx.x / (NTOK / 128);  // 96 channel blocks
    const int t_base = bm * 128;
    const int o_base = bn * 128;

    v8f c[2][4] = {};
    gemm_tiles<DIM, DIM>(feat + (long)t_base * DIM,
                         w    + (long)o_base * DIM, sA, sB, c);

    const int lane = threadIdx.x & 31;
    const int wid  = threadIdx.x >> 5;
    const int wm   = wid & 3, wn = wid >> 2;
    const int kh   = lane >> 4;

    #pragma unroll
    for (int j = 0; j < 4; ++j) {
        const int   o     = o_base + wn * 64 + j * 16 + (lane & 15);
        const float biasv = b_ff[o];
        const int   se    = o & 1;
        const int   d     = (o >> 1) & (LDIM - 1);
        const int   label = o >> 9;
        bf16_t* dstbuf = se ? end_buf : start_buf;
        #pragma unroll
        for (int i = 0; i < 2; ++i) {
            #pragma unroll
            for (int v = 0; v < 8; ++v) {
                const int t  = t_base + wm * 32 + i * 16 + v + 8 * kh;
                const int bb = t >> 9;              // batch
                const int s  = t & (LL - 1);        // position
                float val = c[i][j][v] + biasv;
                val = val > 0.0f ? val : 0.0f;
                dstbuf[(((long)(bb * NLAB + label) * LL + s) * LDIM) + d] = (bf16_t)val;
            }
        }
    }
}

// ---------------------------------------------------------------------------
// Stage 2: biaffine  scores[b,l,s,e] = start[b,l,s,:] . end[b,l,e,:] + bias[l]
// Per (b,label) a 512x512 output in 4x4 blocks of 128x128, K=256.
// Fused epilogue writes both `scores` and `masked_scores`.
// ---------------------------------------------------------------------------
__global__ __launch_bounds__(256)
void biaffine_wmma(const bf16_t* __restrict__ start_buf,
                   const bf16_t* __restrict__ end_buf,
                   const float*  __restrict__ label_bias,  // [NLAB]
                   const unsigned char* __restrict__ mask, // [NB, LL] bool
                   float* __restrict__ scores,             // [NB,NLAB,LL,LL]
                   float* __restrict__ masked)             // [NB,NLAB,LL,LL]
{
    __shared__ bf16_t sA[2][128 * LDS_PITCH];
    __shared__ bf16_t sB[2][128 * LDS_PITCH];

    const int bl    = blockIdx.x >> 4;        // b*NLAB + label
    const int tile  = blockIdx.x & 15;        // 4x4 grid of 128x128
    const int s_blk = (tile >> 2) * 128;
    const int e_blk = (tile & 3) * 128;
    const int label = bl % NLAB;
    const int b     = bl / NLAB;

    const long base = (long)bl * LL * LDIM;

    v8f c[2][4] = {};
    gemm_tiles<LDIM, LDIM>(start_buf + base + (long)s_blk * LDIM,
                           end_buf   + base + (long)e_blk * LDIM, sA, sB, c);

    const int lane = threadIdx.x & 31;
    const int wid  = threadIdx.x >> 5;
    const int wm   = wid & 3, wn = wid >> 2;
    const int kh   = lane >> 4;

    const float lb    = label_bias[label];
    const long  obase = (long)bl * LL * LL;

    #pragma unroll
    for (int j = 0; j < 4; ++j) {
        const int  e  = e_blk + wn * 64 + j * 16 + (lane & 15);
        const bool me = mask[b * LL + e] != 0;
        #pragma unroll
        for (int i = 0; i < 2; ++i) {
            #pragma unroll
            for (int v = 0; v < 8; ++v) {
                const int s = s_blk + wm * 32 + i * 16 + v + 8 * kh;
                float val = c[i][j][v] + lb;
                if (!__builtin_isfinite(val)) val = NEGV;   // remove_nan branch
                const long idx = obase + (long)s * LL + e;
                scores[idx] = val;
                const bool ms = mask[b * LL + s] != 0;
                masked[idx] = (s <= e && ms && me) ? val : NEGV;
            }
        }
    }
}

// ---------------------------------------------------------------------------
// launch
// ---------------------------------------------------------------------------
extern "C" void kernel_launch(void* const* d_in, const int* in_sizes, int n_in,
                              void* d_out, int out_size, void* d_ws, size_t ws_size,
                              hipStream_t stream)
{
    const float*         features = (const float*)d_in[0];         // [8,512,1024]
    const unsigned char* mask     = (const unsigned char*)d_in[1]; // [8,512] bool
    const float*         w_ff     = (const float*)d_in[2];         // [12288,1024]
    const float*         b_ff     = (const float*)d_in[3];         // [12288]
    const float*         bias     = (const float*)d_in[4];         // [24]

    // workspace layout (bf16): feat 8MB | w 24MB | start 48MB | end 48MB
    char*   ws        = (char*)d_ws;
    bf16_t* feat_bf   = (bf16_t*)ws;
    bf16_t* w_bf      = (bf16_t*)(ws + (size_t)NTOK * DIM * 2);
    bf16_t* start_buf = (bf16_t*)(ws + (size_t)(NTOK + OUTCH) * DIM * 2);
    bf16_t* end_buf   = start_buf + (size_t)NB * NLAB * LL * LDIM;

    const long nf = (long)NTOK  * DIM;   // 4,194,304
    const long nw = (long)OUTCH * DIM;   // 12,582,912
    cvt_f32_to_bf16<<<(int)((nf / 4 + 255) / 256), 256, 0, stream>>>(features, feat_bf, nf);
    cvt_f32_to_bf16<<<(int)((nw / 4 + 255) / 256), 256, 0, stream>>>(w_ff, w_bf, nw);

    // FFN: (4096/128) * (12288/128) = 32 * 96 = 3072 blocks
    ffn_wmma<<<3072, 256, 0, stream>>>(feat_bf, w_bf, b_ff, start_buf, end_buf);

    // Biaffine: 192 (b,label) * 16 block tiles = 3072 blocks
    float* scores = (float*)d_out;
    float* masked = scores + (size_t)NB * NLAB * LL * LL;
    biaffine_wmma<<<3072, 256, 0, stream>>>(start_buf, end_buf, bias, mask,
                                            scores, masked);
}
